// AttentionHead_64544768524906
// MI455X (gfx1250) — matrix operand order repaired and checked by reading, not compiled, not run on hardware
//
#include <hip/hip_runtime.h>
#include <hip/hip_bf16.h>

// Problem constants (from reference)
#define B_   256
#define T_   200
#define DM   2048
#define DK   128
#define TQP  208   // 13 * 16 : padded T for Q/K time tiles
#define TVP  224   // 7  * 32 : padded T for P@V contraction

typedef __attribute__((ext_vector_type(8)))  __bf16 v8bf;
typedef __attribute__((ext_vector_type(16))) __bf16 v16bf;
typedef __attribute__((ext_vector_type(8)))  float  v8f;

// ---- fragment loaders (contiguous k = 16h + s in A and B alike, 16-bit A 16x32 / B 32x16) ----
// A: per lane one contiguous 16-elem K run at +0 (lanes 0-15) or +16 (lanes 16-31)
__device__ __forceinline__ v16bf load_afrag(const __bf16* row, int kb, bool hi) {
  return *(const v16bf*)(row + kb + (hi ? 16 : 0));
}
// B: per lane one contiguous 16-elem K run at +0 (lanes 0-15) or +16 (lanes 16-31)
__device__ __forceinline__ v16bf load_bfrag(const __bf16* row, int kb, bool hi) {
  return *(const v16bf*)(row + kb + (hi ? 16 : 0));
}

// ---------------------------------------------------------------------------
// Kernel 0: convert weights fp32 -> bf16 once.  Wb layout [3][DK][DM].
// ---------------------------------------------------------------------------
__global__ __launch_bounds__(256)
void wconv_kernel(const float* __restrict__ Wq, const float* __restrict__ Wk,
                  const float* __restrict__ Wv, __bf16* __restrict__ Wb) {
  const int i = blockIdx.x * 256 + threadIdx.x;      // over DK*DM
  const int N = DK * DM;
  Wb[i]         = (__bf16)Wq[i];
  Wb[N + i]     = (__bf16)Wk[i];
  Wb[2 * N + i] = (__bf16)Wv[i];
}

// ---------------------------------------------------------------------------
// Kernel 1: fused QKV projection, double-buffered LDS staging.
//   One block per 16-row M-tile of flattened q [B*T, DM]; 8 waves each own
//   3 of the 24 N-tiles (Q|K|V x 8 col tiles).  Per 32-wide k-step the block
//   stages the 16x32 q chunk into LDS as bf16 (1 packed cvt per thread);
//   the next chunk's global load is issued before the current WMMA compute
//   so HBM latency overlaps with matrix work (one barrier per k-step).
//   Q,K stored [B, TQP, DK] bf16; V stored transposed [B, DK, TVP] bf16.
// ---------------------------------------------------------------------------
__global__ __launch_bounds__(256)
void qkv_proj_kernel(const float* __restrict__ q,
                     const __bf16* __restrict__ Wb,
                     __bf16* __restrict__ Qb,
                     __bf16* __restrict__ Kb,
                     __bf16* __restrict__ Vt) {
  __shared__ __bf16 As[2][16][32];

  const int wave = threadIdx.x >> 5;
  const int lane = threadIdx.x & 31;
  const int lh   = lane & 15;
  const bool hi  = lane >= 16;

  const int mTile = blockIdx.x;                      // 3200 tiles over B*T

  // staging coords: each thread converts one float2 of the 16x32 chunk
  const int se   = threadIdx.x * 2;                  // 0..510
  const int srow = se >> 5;
  const int scol = se & 31;
  const float* sptr = q + (size_t)(mTile * 16 + srow) * DM + scol;

  // per-wave B rows for its 3 (mat,nTile) pairs; Wb row = mat*DK + nTile*16+lh
  const __bf16* brow[3];
  #pragma unroll
  for (int j = 0; j < 3; ++j) {
    const int sel = wave * 3 + j;                    // 0..23
    const int mat = sel >> 3;
    const int nT  = sel & 7;
    brow[j] = Wb + ((size_t)mat * DK + nT * 16 + lh) * DM;
  }

  v8f acc0 = {}, acc1 = {}, acc2 = {};

  // prologue: stage k=0 into buffer 0
  {
    float2 v = *(const float2*)(sptr);
    As[0][srow][scol]     = (__bf16)v.x;
    As[0][srow][scol + 1] = (__bf16)v.y;
  }
  __syncthreads();

  for (int kb = 0; kb < DM; kb += 64) {
    // ---- step A: compute k=kb from buf0; stage k=kb+32 into buf1 ------
    {
      float2 nxt = *(const float2*)(sptr + kb + 32);       // kb+32 <= 2016 < DM always
      const v16bf a  = load_afrag(&As[0][lh][0], 0, hi);
      const v16bf b0 = load_bfrag(brow[0], kb, hi);
      const v16bf b1 = load_bfrag(brow[1], kb, hi);
      const v16bf b2 = load_bfrag(brow[2], kb, hi);
      acc0 = __builtin_amdgcn_wmma_f32_16x16x32_bf16(false, a, false, b0, (short)0, acc0, false, false);
      acc1 = __builtin_amdgcn_wmma_f32_16x16x32_bf16(false, a, false, b1, (short)0, acc1, false, false);
      acc2 = __builtin_amdgcn_wmma_f32_16x16x32_bf16(false, a, false, b2, (short)0, acc2, false, false);
      As[1][srow][scol]     = (__bf16)nxt.x;
      As[1][srow][scol + 1] = (__bf16)nxt.y;
    }
    __syncthreads();
    // ---- step B: compute k=kb+32 from buf1; stage k=kb+64 into buf0 ---
    {
      const bool more = (kb + 64) < DM;
      float2 nxt = more ? *(const float2*)(sptr + kb + 64) : float2{0.f, 0.f};
      const int k2 = kb + 32;
      const v16bf a  = load_afrag(&As[1][lh][0], 0, hi);
      const v16bf b0 = load_bfrag(brow[0], k2, hi);
      const v16bf b1 = load_bfrag(brow[1], k2, hi);
      const v16bf b2 = load_bfrag(brow[2], k2, hi);
      acc0 = __builtin_amdgcn_wmma_f32_16x16x32_bf16(false, a, false, b0, (short)0, acc0, false, false);
      acc1 = __builtin_amdgcn_wmma_f32_16x16x32_bf16(false, a, false, b1, (short)0, acc1, false, false);
      acc2 = __builtin_amdgcn_wmma_f32_16x16x32_bf16(false, a, false, b2, (short)0, acc2, false, false);
      if (more) {
        As[0][srow][scol]     = (__bf16)nxt.x;
        As[0][srow][scol + 1] = (__bf16)nxt.y;
      }
    }
    __syncthreads();
  }

  // store: C/D layout VGPR r -> row r (+8 for upper lane half), col = lane&15
  #pragma unroll
  for (int j = 0; j < 3; ++j) {
    const int sel = wave * 3 + j;
    const int mat = sel >> 3;
    const int nT  = sel & 7;
    const int col = nT * 16 + lh;
    const v8f& acc = (j == 0) ? acc0 : (j == 1) ? acc1 : acc2;
    #pragma unroll
    for (int r = 0; r < 8; ++r) {
      const int gr = mTile * 16 + r + (hi ? 8 : 0);  // global row in [0, B*T)
      const int bb = gr / T_;
      const int t  = gr - bb * T_;
      const __bf16 val = (__bf16)acc[r];
      if (mat == 0)      Qb[((size_t)bb * TQP + t) * DK + col] = val;
      else if (mat == 1) Kb[((size_t)bb * TQP + t) * DK + col] = val;
      else               Vt[((size_t)bb * DK + col) * TVP + t] = val;
    }
  }
}

// ---------------------------------------------------------------------------
// Kernel 2: attention.  One block per (batch, 16-query tile), 8 waves.
//   Phase A: S = Q @ K^T (WMMA, Q fragments hoisted) -> LDS fp32 [16 x 208]
//   Phase B: masked softmax per row (wave32 shuffle reductions) -> LDS bf16 P
//   Phase C: out = P @ V (WMMA, V read transposed from workspace)
// ---------------------------------------------------------------------------
__global__ __launch_bounds__(256)
void attn_kernel(const __bf16* __restrict__ Qb,
                 const __bf16* __restrict__ Kb,
                 const __bf16* __restrict__ Vt,
                 const int* __restrict__ pad_mask,
                 float* __restrict__ out) {
  __shared__ float  S[16][TQP];
  __shared__ __bf16 P[16][TVP];
  __shared__ int    padm[TVP];

  const int b    = blockIdx.y;
  const int q0   = blockIdx.x * 16;
  const int wave = threadIdx.x >> 5;
  const int lane = threadIdx.x & 31;
  const int lh   = lane & 15;
  const bool hi  = lane >= 16;

  for (int t = threadIdx.x; t < TVP; t += 256)
    padm[t] = (t < T_) ? pad_mask[b * T_ + t] : 1;

  // ---- Phase A: score tiles S[16 x 208] -------------------------------
  const __bf16* qrow = Qb + ((size_t)b * TQP + q0 + lh) * DK;
  v16bf af[4];
  #pragma unroll
  for (int kk = 0; kk < 4; ++kk) af[kk] = load_afrag(qrow, kk * 32, hi);

  for (int nt = wave; nt < 13; nt += 8) {
    const __bf16* krow = Kb + ((size_t)b * TQP + nt * 16 + lh) * DK;
    v8f acc = {};
    #pragma unroll
    for (int kk = 0; kk < 4; ++kk) {
      const v16bf bb = load_bfrag(krow, kk * 32, hi);
      acc = __builtin_amdgcn_wmma_f32_16x16x32_bf16(false, af[kk], false, bb, (short)0, acc, false, false);
    }
    #pragma unroll
    for (int r = 0; r < 8; ++r)
      S[r + (hi ? 8 : 0)][nt * 16 + lh] = acc[r];
  }
  __syncthreads();

  // ---- Phase B: masked softmax (2 rows per wave, lane-parallel) -------
  const float scale = 0.022097086912079608f;  // 1/sqrt(2048) (ref scales by d_model)
  const float NEG_INF = -__builtin_inff();
  #pragma unroll
  for (int rr = 0; rr < 2; ++rr) {
    const int r  = wave * 2 + rr;
    const int qg = q0 + r;
    float sc[7];
    float mx = NEG_INF;
    #pragma unroll
    for (int i = 0; i < 7; ++i) {
      const int t = lane + i * 32;
      const bool valid = (t < T_) && (t <= qg) && (padm[t] == 0);
      const float v = (valid && t < TQP) ? S[r][t] * scale : NEG_INF;
      sc[i] = v;
      mx = fmaxf(mx, v);
    }
    #pragma unroll
    for (int off = 16; off > 0; off >>= 1)
      mx = fmaxf(mx, __shfl_xor(mx, off, 32));
    float sum = 0.f;
    #pragma unroll
    for (int i = 0; i < 7; ++i) {
      const float e = (sc[i] == NEG_INF) ? 0.f : __expf(sc[i] - mx);
      sc[i] = e;
      sum += e;
    }
    #pragma unroll
    for (int off = 16; off > 0; off >>= 1)
      sum += __shfl_xor(sum, off, 32);
    const float inv = 1.0f / sum;   // >=1 valid key guaranteed by reference
    #pragma unroll
    for (int i = 0; i < 7; ++i)
      P[r][lane + i * 32] = (__bf16)(sc[i] * inv);
  }
  __syncthreads();

  // ---- Phase C: out = P @ V (one 16x16 output tile per wave) ----------
  const int n0 = wave * 16;
  const __bf16* vrow = Vt + ((size_t)b * DK + n0 + lh) * TVP;
  const __bf16* prow = &P[lh][0];
  v8f acc = {};
  #pragma unroll
  for (int kb = 0; kb < TVP; kb += 32) {
    const v16bf a  = load_afrag(prow, kb, hi);
    const v16bf bb = load_bfrag(vrow, kb, hi);
    acc = __builtin_amdgcn_wmma_f32_16x16x32_bf16(false, a, false, bb, (short)0, acc, false, false);
  }
  #pragma unroll
  for (int r = 0; r < 8; ++r) {
    const int qg = q0 + r + (hi ? 8 : 0);
    if (qg < T_)
      out[((size_t)b * T_ + qg) * DK + n0 + lh] = acc[r];
  }
}

// ---------------------------------------------------------------------------
extern "C" void kernel_launch(void* const* d_in, const int* in_sizes, int n_in,
                              void* d_out, int out_size, void* d_ws, size_t ws_size,
                              hipStream_t stream) {
  const float* q  = (const float*)d_in[0];
  const int*   pm = (const int*)d_in[1];
  const float* Wq = (const float*)d_in[2];
  const float* Wk = (const float*)d_in[3];
  const float* Wv = (const float*)d_in[4];
  float* out = (float*)d_out;

  __bf16* Qb = (__bf16*)d_ws;                         // B*TQP*DK bf16 (13.6 MB)
  __bf16* Kb = Qb + (size_t)B_ * TQP * DK;            // B*TQP*DK bf16 (13.6 MB)
  __bf16* Vt = Kb + (size_t)B_ * TQP * DK;            // B*DK*TVP bf16 (14.7 MB)
  __bf16* Wb = Vt + (size_t)B_ * DK * TVP;            // 3*DK*DM  bf16 ( 1.5 MB)

  wconv_kernel<<<(DK * DM) / 256, 256, 0, stream>>>(Wq, Wk, Wv, Wb);
  qkv_proj_kernel<<<3200, 256, 0, stream>>>(q, Wb, Qb, Kb, Vt);
  dim3 grid(13, B_);
  attn_kernel<<<grid, 256, 0, stream>>>(Qb, Kb, Vt, pm, out);
}
